// ggm_35442070127266
// MI455X (gfx1250) — matrix operand order repaired and checked by reading, not compiled, not run on hardware
//
#include <hip/hip_runtime.h>
#include <hip/hip_bf16.h>
#include <math.h>

typedef __bf16 bf16;
typedef __attribute__((ext_vector_type(16))) __bf16 v16bf;
typedef __attribute__((ext_vector_type(8)))  __bf16 v8bf;
typedef __attribute__((ext_vector_type(8)))  float  v8f;
typedef unsigned long long u64;

#define NNODES 100000
#define NEDGES 400000
#define D      128
#define DE     64
#define KIN    322   // U_W row length
#define K320   320   // WMMA-covered K (cond folded into bias)
#define XPITCH 336   // LDS row pitch (bf16) for edge x tile: 672B = 42*16B
#define HPITCH 136   // LDS row pitch (bf16) for 128-wide tiles: 272B = 17*16B

// ---------------- WMMA helpers ----------------

__device__ __forceinline__ v8f wmma_bf16(v16bf a, v16bf b, v8f c) {
  // v_wmma_f32_16x16x32_bf16: D = A(16x32) * B(32x16) + C
  return __builtin_amdgcn_wmma_f32_16x16x32_bf16(false, a, false, b, (short)0, c, false, false);
}

// A fragment: lane holds 16 contiguous K starting at hi*16
// p must point at row_base + kstep*32 + hi*16 (16B aligned)
__device__ __forceinline__ v16bf ldfragA(const bf16* p) {
  v16bf r;
  ((v8bf*)&r)[0] = *(const v8bf*)(p);
  ((v8bf*)&r)[1] = *(const v8bf*)(p + 8);
  return r;
}

// B fragment (weights stored [N][K] row-major): lane holds 16 contiguous K
// starting at hi*16. p must point at wrow_base + kstep*32 + hi*16 (32B aligned)
__device__ __forceinline__ v16bf ldfragB(const bf16* p) {
  v16bf r;
  ((v8bf*)&r)[0] = *(const v8bf*)(p);
  ((v8bf*)&r)[1] = *(const v8bf*)(p + 8);
  return r;
}

// ---------------- prep kernels ----------------

__global__ void cvt_kernel(const float* __restrict__ s, bf16* __restrict__ d, int n) {
  int i = blockIdx.x * blockDim.x + threadIdx.x;
  if (i < n) d[i] = (bf16)s[i];
}

// U_W is [3*128][322]; keep only first 320 cols, bf16 row-major [3*128][320]
__global__ void cvt_uw_kernel(const float* __restrict__ s, bf16* __restrict__ d, int n) {
  int i = blockIdx.x * blockDim.x + threadIdx.x;
  if (i < n) {
    int r = i / K320, c = i - r * K320;
    d[i] = (bf16)s[(size_t)r * KIN + c];
  }
}

// bcond[k*128+n] = U_b + U_W[:,320]*cond0 + U_W[:,321]*cond1
__global__ void bias_kernel(const float* __restrict__ uw, const float* __restrict__ ub,
                            const float* __restrict__ cond, float* __restrict__ bc) {
  int i = blockIdx.x * blockDim.x + threadIdx.x;
  if (i < 3 * D)
    bc[i] = ub[i] + uw[(size_t)i * KIN + 320] * cond[0] + uw[(size_t)i * KIN + 321] * cond[1];
}

__global__ void zero_kernel(float4* __restrict__ p, int n4) {
  int i = blockIdx.x * blockDim.x + threadIdx.x;
  if (i < n4) p[i] = make_float4(0.f, 0.f, 0.f, 0.f);
}

// ---------------- edge message + scatter-add ----------------
// block = 128 threads (4 waves); 64 edges per block, 16 edges per wave.
// msg[16x128] = x[16x320] @ W^T, x = [h_dst | h_src | e] in bf16 LDS.
__global__ __launch_bounds__(128) void edge_kernel(
    const float* __restrict__ h_in, const float* __restrict__ efeat,
    const int* __restrict__ srci, const int* __restrict__ dsti,
    const bf16* __restrict__ wU,   // [128][320] bf16 (round base)
    const float* __restrict__ bvec, // [128] f32 (round base, includes cond)
    float* __restrict__ agg) {
  __shared__ __align__(16) bf16 xs[64 * XPITCH];
  __shared__ int sd[64];
  __shared__ int ss[64];

  const int tid = threadIdx.x;
  const int e0 = blockIdx.x * 64;

  if (tid < 64) sd[tid] = dsti[e0 + tid];
  else          ss[tid - 64] = srci[e0 + tid - 64];
  __syncthreads();

  // gather + f32->bf16: 64 rows x 80 float4
  for (int t = tid; t < 64 * 80; t += 128) {
    int r = t / 80, q = t - r * 80;
    int col = q * 4;
    float4 v;
    if (col < 128)      v = *(const float4*)&h_in[(size_t)sd[r] * D + col];
    else if (col < 256) v = *(const float4*)&h_in[(size_t)ss[r] * D + (col - 128)];
    else                v = *(const float4*)&efeat[(size_t)(e0 + r) * DE + (col - 256)];
    bf16 tmp[4] = {(bf16)v.x, (bf16)v.y, (bf16)v.z, (bf16)v.w};
    *(u64*)&xs[r * XPITCH + col] = *(const u64*)tmp;
  }
  __syncthreads();

  const int w = tid >> 5, lane = tid & 31;
  const int ln = lane & 15, hi = lane >> 4;
  const int rl = w * 16 + ln;

  v8f acc[8];
#pragma unroll
  for (int nt = 0; nt < 8; ++nt) acc[nt] = (v8f){0.f,0.f,0.f,0.f,0.f,0.f,0.f,0.f};

  for (int kk = 0; kk < 10; ++kk) {          // K = 320 = 10*32
    v16bf a = ldfragA(&xs[rl * XPITCH + kk * 32 + hi * 16]);
#pragma unroll
    for (int nt = 0; nt < 8; ++nt) {         // N = 128 = 8*16
      v16bf b = ldfragB(&wU[(size_t)(nt * 16 + ln) * K320 + kk * 32 + hi * 16]);
      acc[nt] = wmma_bf16(a, b, acc[nt]);
    }
  }

  // scatter segment-sum: each (edge, channel) contributes msg + bias
#pragma unroll
  for (int nt = 0; nt < 8; ++nt) {
    const int c = nt * 16 + ln;
    const float bias = bvec[c];
#pragma unroll
    for (int j = 0; j < 8; ++j) {
      const int m = hi * 8 + j;              // edge row within wave tile
      const int dn = sd[w * 16 + m];
      __hip_atomic_fetch_add(&agg[(size_t)dn * D + c], acc[nt][j] + bias,
                             __ATOMIC_RELAXED, __HIP_MEMORY_SCOPE_AGENT);
    }
  }
}

// ---------------- GRU cell ----------------
// block = 128 threads (4 waves); 16 nodes per block; wave w owns channels
// [w*32, w*32+32) for all 3 gates of both matmuls (K = 128 = 4*32).
__global__ __launch_bounds__(128) void gru_kernel(
    const float* __restrict__ agg, const float* __restrict__ h_in,
    const bf16* __restrict__ wih, const bf16* __restrict__ whh, // [384][128] bf16
    const float* __restrict__ bih, const float* __restrict__ bhh, // [384] f32
    float* __restrict__ h_out) {
  __shared__ __align__(16) bf16 as_[16 * HPITCH];
  __shared__ __align__(16) bf16 hs_[16 * HPITCH];

  const int tid = threadIdx.x;
  const size_t n0 = (size_t)blockIdx.x * 16;

  for (int t = tid; t < 1024; t += 128) {    // 2 arrays * 16 rows * 32 float4
    int which = t >> 9;
    int rem = t & 511;
    int r = rem >> 5, q = rem & 31;
    int col = q * 4;
    const float* sp = which ? h_in : agg;
    float4 v = *(const float4*)&sp[(n0 + r) * D + col];
    bf16 tmp[4] = {(bf16)v.x, (bf16)v.y, (bf16)v.z, (bf16)v.w};
    bf16* dp = which ? hs_ : as_;
    *(u64*)&dp[r * HPITCH + col] = *(const u64*)tmp;
  }
  __syncthreads();

  const int w = tid >> 5, lane = tid & 31;
  const int ln = lane & 15, hi = lane >> 4;
  const int cb = w * 32;

  v8f gi[3][2], gh[3][2];
#pragma unroll
  for (int g = 0; g < 3; ++g)
#pragma unroll
    for (int nt = 0; nt < 2; ++nt) {
      gi[g][nt] = (v8f){0.f,0.f,0.f,0.f,0.f,0.f,0.f,0.f};
      gh[g][nt] = (v8f){0.f,0.f,0.f,0.f,0.f,0.f,0.f,0.f};
    }

  for (int kk = 0; kk < 4; ++kk) {           // K = 128
    v16bf aA = ldfragA(&as_[ln * HPITCH + kk * 32 + hi * 16]);
    v16bf aH = ldfragA(&hs_[ln * HPITCH + kk * 32 + hi * 16]);
#pragma unroll
    for (int g = 0; g < 3; ++g) {
#pragma unroll
      for (int nt = 0; nt < 2; ++nt) {
        const size_t nrow = (size_t)(g * D + cb + nt * 16 + ln);
        v16bf bI = ldfragB(&wih[nrow * D + kk * 32 + hi * 16]);
        gi[g][nt] = wmma_bf16(aA, bI, gi[g][nt]);
        v16bf bH = ldfragB(&whh[nrow * D + kk * 32 + hi * 16]);
        gh[g][nt] = wmma_bf16(aH, bH, gh[g][nt]);
      }
    }
  }

  // gates in f32; h read from global at full precision
#pragma unroll
  for (int nt = 0; nt < 2; ++nt) {
    const int c = cb + nt * 16 + ln;
    const float bi_r = bih[c], bi_z = bih[D + c], bi_n = bih[2 * D + c];
    const float bh_r = bhh[c], bh_z = bhh[D + c], bh_n = bhh[2 * D + c];
#pragma unroll
    for (int j = 0; j < 8; ++j) {
      const int m = hi * 8 + j;
      const size_t idx = (n0 + m) * D + c;
      const float ir  = gi[0][nt][j] + bi_r;
      const float iz  = gi[1][nt][j] + bi_z;
      const float inn = gi[2][nt][j] + bi_n;
      const float hr  = gh[0][nt][j] + bh_r;
      const float hz  = gh[1][nt][j] + bh_z;
      const float hn  = gh[2][nt][j] + bh_n;
      const float r = 1.f / (1.f + __expf(-(ir + hr)));
      const float z = 1.f / (1.f + __expf(-(iz + hz)));
      const float nn = tanhf(inn + r * hn);
      const float hv = h_in[idx];
      h_out[idx] = (1.f - z) * nn + z * hv;
    }
  }
}

// ---------------- launch ----------------

extern "C" void kernel_launch(void* const* d_in, const int* in_sizes, int n_in,
                              void* d_out, int out_size, void* d_ws, size_t ws_size,
                              hipStream_t stream) {
  const float* h0    = (const float*)d_in[0];
  const float* efeat = (const float*)d_in[1];
  const int*   srci  = (const int*)d_in[2];
  const int*   dsti  = (const int*)d_in[3];
  const float* cond  = (const float*)d_in[4];
  const float* U_W   = (const float*)d_in[5];
  const float* U_b   = (const float*)d_in[6];
  const float* Wih   = (const float*)d_in[7];
  const float* Whh   = (const float*)d_in[8];
  const float* bih   = (const float*)d_in[9];
  const float* bhh   = (const float*)d_in[10];
  float* out = (float*)d_out;

  size_t off = 0;
  auto wsalloc = [&](size_t bytes) -> void* {
    void* p = (char*)d_ws + off;
    off += (bytes + 255) & ~(size_t)255;
    return p;
  };
  bf16*  wU_bf  = (bf16*)wsalloc((size_t)3 * D * K320 * sizeof(bf16));
  bf16*  wih_bf = (bf16*)wsalloc((size_t)3 * 3 * D * D * sizeof(bf16));
  bf16*  whh_bf = (bf16*)wsalloc((size_t)3 * 3 * D * D * sizeof(bf16));
  float* bcond  = (float*)wsalloc((size_t)3 * D * sizeof(float));
  float* agg    = (float*)wsalloc((size_t)NNODES * D * sizeof(float));
  float* hmid   = (float*)wsalloc((size_t)NNODES * D * sizeof(float));

  // weight conversion + cond folding (tiny, once per launch)
  {
    int n_uw = 3 * D * K320;                 // 122880
    cvt_uw_kernel<<<(n_uw + 255) / 256, 256, 0, stream>>>(U_W, wU_bf, n_uw);
    int n_g = 3 * 3 * D * D;                 // 147456
    cvt_kernel<<<(n_g + 255) / 256, 256, 0, stream>>>(Wih, wih_bf, n_g);
    cvt_kernel<<<(n_g + 255) / 256, 256, 0, stream>>>(Whh, whh_bf, n_g);
    bias_kernel<<<3, 128, 0, stream>>>(U_W, U_b, cond, bcond);
  }

  const float* hin[3]  = {h0, out, hmid};
  float*       hout[3] = {out, hmid, out};

  const int n4 = NNODES * D / 4;
  for (int k = 0; k < 3; ++k) {
    zero_kernel<<<(n4 + 255) / 256, 256, 0, stream>>>((float4*)agg, n4);
    edge_kernel<<<NEDGES / 64, 128, 0, stream>>>(
        hin[k], efeat, srci, dsti,
        wU_bf + (size_t)k * D * K320, bcond + (size_t)k * D, agg);
    gru_kernel<<<NNODES / 16, 128, 0, stream>>>(
        agg, hin[k],
        wih_bf + (size_t)k * 3 * D * D, whh_bf + (size_t)k * 3 * D * D,
        bih + (size_t)k * 3 * D, bhh + (size_t)k * 3 * D, hout[k]);
  }
}
